// SEGCN_29892972380180
// MI455X (gfx1250) — compile-verified, matrix operand order checked by reading, not run on hardware
//
#include <hip/hip_runtime.h>
#include <hip/hip_bf16.h>

// ---------------------------------------------------------------------------
// SEGCN forward for MI455X (gfx1250).
// Heavy GEMMs run on v_wmma_f32_16x16x32_bf16 (wave32). f32 operands are
// converted to packed bf16 on the LDS staging path (v_add + v_perm_b32 pack).
// Block tile 128x64, 8 waves x (32x32 macro-tile) -> 4 WMMAs per wave per
// k-step; branch-free steady-state loop, guarded tail/edge loop.
// ---------------------------------------------------------------------------

typedef __attribute__((ext_vector_type(16))) __bf16 v16bf;
typedef __attribute__((ext_vector_type(8)))  float  v8f;

#define NN 4096

__device__ __forceinline__ unsigned fbits(float f) {
    union { float f; unsigned u; } v; v.f = f;
    return v.u;
}
// round-to-nearest bf16 (ties away): one v_add each + one v_perm_b32 merge
__device__ __forceinline__ unsigned pack2bf(float lo, float hi) {
    unsigned l = fbits(lo) + 0x8000u;
    unsigned h = fbits(hi) + 0x8000u;
    return __builtin_amdgcn_perm(h, l, 0x07060302u);   // {h.b3,h.b2,l.b3,l.b2}
}
__device__ __forceinline__ unsigned short f2bf(float f) {
    return (unsigned short)((fbits(f) + 0x8000u) >> 16);
}

// ---------------------------------------------------------------------------
// C[M,N] = A[M,K] * B (+Cold) (+bias) (relu?)
// A row-major f32. transB==0: B is [K,N] row-major. transB==1: B is [N,K].
// ---------------------------------------------------------------------------
__global__ __launch_bounds__(256)
void gemm_wmma_bf16(const float* __restrict__ A, const float* __restrict__ B,
                    float* __restrict__ C, const float* __restrict__ bias,
                    int M, int N, int K, int transB, int relu, int accum)
{
    // padded stride 40 halves = 80B: b128 frags stay 16B aligned, no conflicts
    __shared__ __align__(16) unsigned short As[128 * 40];
    __shared__ __align__(16) unsigned short Bs[64 * 40];

    const int tid    = threadIdx.x;
    const int m_base = blockIdx.y * 128;
    const int n_base = blockIdx.x * 64;
    const int wid    = tid >> 5;
    const int lane   = tid & 31;
    const int lrow   = lane & 15;
    const int g      = lane >> 4;            // half-wave group
    const int m_off  = (wid >> 1) * 32;      // 4 wave-rows, 32 tall
    const int n_off  = (wid & 1) * 32;       // 2 wave-cols, 32 wide

    // staging coords (pair-granular): rows x 16 k-pairs
    const int pr  = tid >> 4;                // 0..15
    const int pcp = tid & 15;                // k-pair index 0..15

    v8f acc00 = {}, acc01 = {}, acc10 = {}, acc11 = {};

    // ISA 7.12.2 16-bit operand layout: lane halves 0..7 = K g*8..+7,
    // halves 8..15 = K 16+g*8..+7  -> two b128 LDS loads per fragment.
    auto compute = [&]() {
        union { v16bf v; uint4 q[2]; } fa0, fa1, fb0, fb1;
        fa0.q[0] = *reinterpret_cast<const uint4*>(&As[(m_off + lrow) * 40 + g * 8]);
        fa0.q[1] = *reinterpret_cast<const uint4*>(&As[(m_off + lrow) * 40 + 16 + g * 8]);
        fa1.q[0] = *reinterpret_cast<const uint4*>(&As[(m_off + 16 + lrow) * 40 + g * 8]);
        fa1.q[1] = *reinterpret_cast<const uint4*>(&As[(m_off + 16 + lrow) * 40 + 16 + g * 8]);
        fb0.q[0] = *reinterpret_cast<const uint4*>(&Bs[(n_off + lrow) * 40 + g * 8]);
        fb0.q[1] = *reinterpret_cast<const uint4*>(&Bs[(n_off + lrow) * 40 + 16 + g * 8]);
        fb1.q[0] = *reinterpret_cast<const uint4*>(&Bs[(n_off + 16 + lrow) * 40 + g * 8]);
        fb1.q[1] = *reinterpret_cast<const uint4*>(&Bs[(n_off + 16 + lrow) * 40 + 16 + g * 8]);
        acc00 = __builtin_amdgcn_wmma_f32_16x16x32_bf16(
                    false, fa0.v, false, fb0.v, (short)0, acc00, false, false);
        acc01 = __builtin_amdgcn_wmma_f32_16x16x32_bf16(
                    false, fa0.v, false, fb1.v, (short)0, acc01, false, false);
        acc10 = __builtin_amdgcn_wmma_f32_16x16x32_bf16(
                    false, fa1.v, false, fb0.v, (short)0, acc10, false, false);
        acc11 = __builtin_amdgcn_wmma_f32_16x16x32_bf16(
                    false, fa1.v, false, fb1.v, (short)0, acc11, false, false);
    };

    const bool fullMN = (m_base + 128 <= M) && (n_base + 64 <= N);
    const int  kMain  = fullMN ? (K & ~31) : 0;

    int k0 = 0;
    // ---- branch-free steady-state loop ----
    for (; k0 < kMain; k0 += 32) {
#pragma unroll
        for (int i = 0; i < 8; ++i) {            // A tile 128x32
            int r = pr + i * 16;
            const float2 v = *(const float2*)(A + (long)(m_base + r) * K + k0 + pcp * 2);
            *(unsigned*)&As[r * 40 + pcp * 2] = pack2bf(v.x, v.y);
        }
        if (transB) {
#pragma unroll
            for (int i = 0; i < 4; ++i) {        // B tile 64x32 (B is [N,K])
                int r = pr + i * 16;
                const float2 v = *(const float2*)(B + (long)(n_base + r) * K + k0 + pcp * 2);
                *(unsigned*)&Bs[r * 40 + pcp * 2] = pack2bf(v.x, v.y);
            }
        } else {
#pragma unroll
            for (int i = 0; i < 4; ++i) {        // B tile 64x32 (B is [K,N])
                int r = pr + i * 16;
                const float* p = B + (long)(k0 + pcp * 2) * N + n_base + r;
                *(unsigned*)&Bs[r * 40 + pcp * 2] = pack2bf(p[0], p[N]);
            }
        }
        __syncthreads();
        compute();
        __syncthreads();
    }
    // ---- guarded tail / edge-tile loop ----
    for (; k0 < K; k0 += 32) {
#pragma unroll
        for (int i = 0; i < 16; ++i) {           // A tile 128x32, element-wise
            int e = tid + i * 256;
            int r = e >> 5, c = e & 31;
            int gm = m_base + r, gk = k0 + c;
            float v = (gm < M && gk < K) ? A[(long)gm * K + gk] : 0.0f;
            As[r * 40 + c] = f2bf(v);
        }
#pragma unroll
        for (int i = 0; i < 8; ++i) {            // B tile 64x32, element-wise
            int e = tid + i * 256;
            int r = e >> 5, c = e & 31;          // r = n, c = k
            int gn = n_base + r, gk = k0 + c;
            float v = 0.0f;
            if (gn < N && gk < K)
                v = transB ? B[(long)gn * K + gk] : B[(long)gk * N + gn];
            Bs[r * 40 + c] = f2bf(v);
        }
        __syncthreads();
        compute();
        __syncthreads();
    }

    // ---- epilogue: C layout VGPR r -> M = r + 8*g, N = lane%16 ----
    auto epi = [&](const v8f& a, int row0, int col) {
        if (col >= N) return;
#pragma unroll
        for (int r = 0; r < 8; ++r) {
            int row = row0 + r + 8 * g;
            if (row < M) {
                long idx = (long)row * N + col;
                float v = a[r];
                if (accum) v += C[idx];
                if (bias)  v += bias[col];
                if (relu)  v = fmaxf(v, 0.0f);
                C[idx] = v;
            }
        }
    };
    const int col0 = n_base + n_off + lrow;
    epi(acc00, m_base + m_off,      col0);
    epi(acc01, m_base + m_off,      col0 + 16);
    epi(acc10, m_base + m_off + 16, col0);
    epi(acc11, m_base + m_off + 16, col0 + 16);
}

// ---------------------------------------------------------------------------
// Row softmax (one block per row, any #cols). Safe in-place.
// ---------------------------------------------------------------------------
__global__ __launch_bounds__(256)
void row_softmax(const float* __restrict__ src, float* __restrict__ dst, int cols)
{
    __shared__ float red[256];
    const long row = blockIdx.x;
    const float* s = src + row * cols;
    float m = -3.402823466e38f;
    for (int c = threadIdx.x; c < cols; c += 256) m = fmaxf(m, s[c]);
    red[threadIdx.x] = m; __syncthreads();
    for (int o = 128; o > 0; o >>= 1) {
        if (threadIdx.x < o) red[threadIdx.x] = fmaxf(red[threadIdx.x], red[threadIdx.x + o]);
        __syncthreads();
    }
    m = red[0]; __syncthreads();
    float sum = 0.f;
    for (int c = threadIdx.x; c < cols; c += 256) sum += __expf(s[c] - m);
    red[threadIdx.x] = sum; __syncthreads();
    for (int o = 128; o > 0; o >>= 1) {
        if (threadIdx.x < o) red[threadIdx.x] += red[threadIdx.x + o];
        __syncthreads();
    }
    float inv = 1.0f / red[0];
    float* d = dst + row * cols;
    for (int c = threadIdx.x; c < cols; c += 256) d[c] = __expf(s[c] - m) * inv;
}

// leaky_relu -> softmax over C (<=8) -> L2 row normalize
__global__ void gate_norm(const float* __restrict__ in, float* __restrict__ out,
                          int rows, int C)
{
    int r = blockIdx.x * blockDim.x + threadIdx.x;
    if (r >= rows) return;
    float v[8];
    float m = -3.402823466e38f;
    for (int c = 0; c < C; ++c) {
        float x = in[(long)r * C + c];
        x = x >= 0.f ? x : 0.01f * x;
        v[c] = x; m = fmaxf(m, x);
    }
    float s = 0.f;
    for (int c = 0; c < C; ++c) { v[c] = __expf(v[c] - m); s += v[c]; }
    float inv = 1.0f / s, n2 = 0.f;
    for (int c = 0; c < C; ++c) { v[c] *= inv; n2 += v[c] * v[c]; }
    float invn = 1.0f / fmaxf(sqrtf(n2), 1e-12f);
    for (int c = 0; c < C; ++c) out[(long)r * C + c] = v[c] * invn;
}

__global__ void k_add_scaled(const float* __restrict__ a, const float* __restrict__ b,
                             const float* __restrict__ s, float* __restrict__ o, long n)
{
    long i = (long)blockIdx.x * 256 + threadIdx.x;
    if (i < n) o[i] = a[i] + s[0] * b[i];
}

__global__ void k_axpby(const float* __restrict__ s1, const float* __restrict__ x,
                        const float* __restrict__ s2, const float* __restrict__ y,
                        float* __restrict__ o, long n)
{
    long i = (long)blockIdx.x * 256 + threadIdx.x;
    if (i < n) o[i] = s1[0] * x[i] + s2[0] * y[i];
}

__global__ void k_axpy1(const float* __restrict__ s, const float* __restrict__ x,
                        const float* __restrict__ y, float* __restrict__ o, long n)
{
    long i = (long)blockIdx.x * 256 + threadIdx.x;
    if (i < n) o[i] = s[0] * x[i] + y[i];
}

// o = m[:,0]*z + m[:,1]*h
__global__ void k_gate_mix(const float* __restrict__ m, const float* __restrict__ z,
                           const float* __restrict__ h, float* __restrict__ o,
                           int rows, int D)
{
    long i = (long)blockIdx.x * 256 + threadIdx.x;
    if (i >= (long)rows * D) return;
    long r = i / D;
    o[i] = m[r * 2 + 0] * z[i] + m[r * 2 + 1] * h[i];
}

// net_in = [u0*z1(500) | u1*z2(500) | u2*z3(2000) | u3*z4(10) | u4*z(10)]
__global__ void k_net_in(const float* __restrict__ u, const float* __restrict__ z1,
                         const float* __restrict__ z2, const float* __restrict__ z3,
                         const float* __restrict__ z4, const float* __restrict__ z,
                         float* __restrict__ o, int rows)
{
    long i = (long)blockIdx.x * 256 + threadIdx.x;
    if (i >= (long)rows * 3020) return;
    long r = i / 3020; int c = (int)(i % 3020);
    const float* ur = u + r * 5;
    float v;
    if      (c < 500)  v = ur[0] * z1[r * 500  + c];
    else if (c < 1000) v = ur[1] * z2[r * 500  + (c - 500)];
    else if (c < 3000) v = ur[2] * z3[r * 2000 + (c - 1000)];
    else if (c < 3010) v = ur[3] * z4[r * 10   + (c - 3000)];
    else               v = ur[4] * z [r * 10   + (c - 3010)];
    o[i] = v;
}

// x_in = [u0*h1(500) | u1*h2(500) | u2*h3(2000) | u3*z(10)]
__global__ void k_x_in(const float* __restrict__ u, const float* __restrict__ h1,
                       const float* __restrict__ h2, const float* __restrict__ h3,
                       const float* __restrict__ z, float* __restrict__ o, int rows)
{
    long i = (long)blockIdx.x * 256 + threadIdx.x;
    if (i >= (long)rows * 3010) return;
    long r = i / 3010; int c = (int)(i % 3010);
    const float* ur = u + r * 5;
    float v;
    if      (c < 500)  v = ur[0] * h1[r * 500  + c];
    else if (c < 1000) v = ur[1] * h2[r * 500  + (c - 500)];
    else if (c < 3000) v = ur[2] * h3[r * 2000 + (c - 1000)];
    else               v = ur[3] * z [r * 10   + (c - 3000)];
    o[i] = v;
}

// Student-t assignment, V=1 -> q_ij = (1+||zq_i-mu_j||^2)^-1, row-normalized
__global__ void k_student_q(const float* __restrict__ zq, const float* __restrict__ cl,
                            float* __restrict__ q, int rows)
{
    int r = blockIdx.x * blockDim.x + threadIdx.x;
    if (r >= rows) return;
    float num[10]; float s = 0.f;
    for (int j = 0; j < 10; ++j) {
        float d2 = 0.f;
        for (int k = 0; k < 10; ++k) {
            float d = zq[(long)r * 10 + k] - cl[j * 10 + k];
            d2 += d * d;
        }
        num[j] = 1.0f / (1.0f + d2);
        s += num[j];
    }
    float inv = 1.0f / s;
    for (int j = 0; j < 10; ++j) q[(long)r * 10 + j] = num[j] * inv;
}

// ---------------------------------------------------------------------------
// Host orchestration
// ---------------------------------------------------------------------------
static inline void gemm(hipStream_t s, const float* A, const float* B, float* C,
                        const float* bias, int M, int N, int K,
                        bool tB, bool relu, bool acc)
{
    dim3 g((N + 63) / 64, (M + 127) / 128);
    gemm_wmma_bf16<<<g, 256, 0, s>>>(A, B, C, bias, M, N, K,
                                     (int)tB, (int)relu, (int)acc);
}

static inline dim3 lin_grid(long n) { return dim3((unsigned)((n + 255) / 256)); }

extern "C" void kernel_launch(void* const* d_in, const int* in_sizes, int n_in,
                              void* d_out, int out_size, void* d_ws, size_t ws_size,
                              hipStream_t stream)
{
    (void)in_sizes; (void)n_in; (void)out_size; (void)ws_size;
    const float* x      = (const float*)d_in[0];
    const float* adj    = (const float*)d_in[1];
    const float* sim    = (const float*)d_in[2];
    const float* enc1_w = (const float*)d_in[3];
    const float* enc1_b = (const float*)d_in[4];
    const float* enc2_w = (const float*)d_in[5];
    const float* enc2_b = (const float*)d_in[6];
    const float* enc3_w = (const float*)d_in[7];
    const float* enc3_b = (const float*)d_in[8];
    const float* zl_w   = (const float*)d_in[9];
    const float* zl_b   = (const float*)d_in[10];
    const float* dec1_w = (const float*)d_in[11];
    const float* dec1_b = (const float*)d_in[12];
    const float* dec2_w = (const float*)d_in[13];
    const float* dec2_b = (const float*)d_in[14];
    const float* dec3_w = (const float*)d_in[15];
    const float* dec3_b = (const float*)d_in[16];
    const float* xbar_w = (const float*)d_in[17];
    const float* xbar_b = (const float*)d_in[18];
    const float* gnn0_w = (const float*)d_in[19];
    const float* gnn1_w = (const float*)d_in[20];
    const float* gnn2_w = (const float*)d_in[21];
    const float* gnn3_w = (const float*)d_in[22];
    const float* gnnz_w = (const float*)d_in[23];
    const float* gnnh_w = (const float*)d_in[24];
    const float* mlp1_w = (const float*)d_in[25];
    const float* mlp1_b = (const float*)d_in[26];
    const float* mlp2_w = (const float*)d_in[27];
    const float* mlp2_b = (const float*)d_in[28];
    const float* mlp3_w = (const float*)d_in[29];
    const float* mlp3_b = (const float*)d_in[30];
    const float* mlpl_w = (const float*)d_in[31];
    const float* mlpl_b = (const float*)d_in[32];
    const float* cluster= (const float*)d_in[33];
    const float* a41    = (const float*)d_in[34];
    const float* a42    = (const float*)d_in[35];
    const float* alpha  = (const float*)d_in[36];
    const float* beta   = (const float*)d_in[37];

    // ---- output slots (flat concat, return order) ----
    float* out    = (float*)d_out;
    float* o_xbar = out;                       // [4096,1000]
    float* o_q    = out + 4096000;             // [4096,10]
    float* o_pred = o_q + 40960;
    float* o_z    = o_pred + 40960;
    float* o_net  = o_z + 40960;
    float* o_z4   = o_net + 40960;
    float* o_simx = o_z4 + 40960;              // [4096,4096]
    float* o_xpred= o_simx + 16777216;
    float* o_simz = o_xpred + 40960;           // [4096,4096]
    float* o_zq   = o_simz + 16777216;

    // ---- workspace bump allocator ----
    char* wp = (char*)d_ws;
    auto alloc = [&](size_t nfl) {
        float* p = (float*)wp;
        wp += ((nfl * sizeof(float) + 255) & ~(size_t)255);
        return p;
    };
    float* h1   = alloc((size_t)NN * 500);
    float* h2   = alloc((size_t)NN * 500);
    float* h3   = alloc((size_t)NN * 2000);
    float* dd1  = alloc((size_t)NN * 2000);
    float* dd2  = alloc((size_t)NN * 500);
    float* dd3  = alloc((size_t)NN * 500);
    float* adjs = alloc((size_t)NN * NN);
    float* S    = alloc((size_t)NN * NN);
    float* z1   = alloc((size_t)NN * 500);
    float* z2   = alloc((size_t)NN * 500);
    float* z3   = alloc((size_t)NN * 2000);
    float* tmp  = alloc((size_t)NN * 2000);
    float* zc   = alloc((size_t)NN * 2000);
    float* glog = alloc((size_t)NN * 2);
    float* mg   = alloc((size_t)NN * 2);
    float* ulog = alloc((size_t)NN * 5);
    float* u    = alloc((size_t)NN * 5);
    float* zai  = alloc((size_t)NN * 10);
    float* zal  = alloc((size_t)NN * 10);
    float* zas  = alloc((size_t)NN * 10);
    float* netin= alloc((size_t)NN * 3020);
    float* xin  = alloc((size_t)NN * 3010);
    float* tsm  = alloc((size_t)NN * 10);
    float* xoutp= alloc((size_t)NN * 10);

    const long nn2 = (long)NN * NN;

    // adj_s = adj + beta * sim
    k_add_scaled<<<lin_grid(nn2), 256, 0, stream>>>(adj, sim, beta, adjs, nn2);

    // ---- autoencoder ----
    gemm(stream, x,   enc1_w, h1,    enc1_b, NN, 500,  1000, false, true,  false);
    gemm(stream, h1,  enc2_w, h2,    enc2_b, NN, 500,  500,  false, true,  false);
    gemm(stream, h2,  enc3_w, h3,    enc3_b, NN, 2000, 500,  false, true,  false);
    gemm(stream, h3,  zl_w,   o_z,   zl_b,   NN, 10,   2000, false, false, false);
    gemm(stream, o_z, dec1_w, dd1,   dec1_b, NN, 2000, 10,   false, true,  false);
    gemm(stream, dd1, dec2_w, dd2,   dec2_b, NN, 500,  2000, false, true,  false);
    gemm(stream, dd2, dec3_w, dd3,   dec3_b, NN, 500,  500,  false, true,  false);
    gemm(stream, dd3, xbar_w, o_xbar,xbar_b, NN, 1000, 500,  false, false, false);

    // ---- multi-scale GCN ----
    gemm(stream, x,   gnn0_w, tmp, nullptr, NN, 500, 1000, false, false, false);
    gemm(stream, adj, tmp,    z1,  nullptr, NN, 500, NN,   false, true,  false);

    gemm(stream, h1, mlp1_w,           glog, mlp1_b, NN, 2, 500, false, false, false);
    gemm(stream, z1, mlp1_w + 500 * 2, glog, nullptr,NN, 2, 500, false, false, true);
    gate_norm<<<lin_grid(NN), 256, 0, stream>>>(glog, mg, NN, 2);
    k_gate_mix<<<lin_grid((long)NN * 500), 256, 0, stream>>>(mg, z1, h1, zc, NN, 500);
    gemm(stream, zc,   gnn1_w, tmp, nullptr, NN, 500, 500, false, false, false);
    gemm(stream, adjs, tmp,    z2,  nullptr, NN, 500, NN,  false, true,  false);

    gemm(stream, h2, mlp2_w,           glog, mlp2_b, NN, 2, 500, false, false, false);
    gemm(stream, z2, mlp2_w + 500 * 2, glog, nullptr,NN, 2, 500, false, false, true);
    gate_norm<<<lin_grid(NN), 256, 0, stream>>>(glog, mg, NN, 2);
    k_gate_mix<<<lin_grid((long)NN * 500), 256, 0, stream>>>(mg, z2, h2, zc, NN, 500);
    gemm(stream, zc,   gnn2_w, tmp, nullptr, NN, 2000, 500, false, false, false);
    gemm(stream, adjs, tmp,    z3,  nullptr, NN, 2000, NN,  false, true,  false);

    gemm(stream, h3, mlp3_w,            glog, mlp3_b, NN, 2, 2000, false, false, false);
    gemm(stream, z3, mlp3_w + 2000 * 2, glog, nullptr,NN, 2, 2000, false, false, true);
    gate_norm<<<lin_grid(NN), 256, 0, stream>>>(glog, mg, NN, 2);
    k_gate_mix<<<lin_grid((long)NN * 2000), 256, 0, stream>>>(mg, z3, h3, zc, NN, 2000);
    gemm(stream, zc,   gnn3_w, tsm,  nullptr, NN, 10, 2000, false, false, false);
    gemm(stream, adjs, tsm,    o_z4, nullptr, NN, 10, NN,   false, true,  false);

    // ---- global similarity self-attention ----
    k_axpby<<<lin_grid((long)NN * 10), 256, 0, stream>>>(a41, o_z4, a42, o_z, zai, (long)NN * 10);
    gemm(stream, adjs, zai, zal, nullptr, NN, 10, NN, false, false, false);
    gemm(stream, zal, zal, S, nullptr, NN, NN, 10, true, false, false);
    row_softmax<<<NN, 256, 0, stream>>>(S, S, NN);
    gemm(stream, S, zal, zas, nullptr, NN, 10, NN, false, false, false);
    k_axpy1<<<lin_grid((long)NN * 10), 256, 0, stream>>>(alpha, zas, zal, o_zq, (long)NN * 10);
    gemm(stream, o_zq, o_zq, o_simz, nullptr, NN, NN, 10, true, false, false);
    row_softmax<<<NN, 256, 0, stream>>>(o_simz, o_simz, NN);

    // ---- layer attention fusion ----
    gemm(stream, z1,   mlpl_w,            ulog, mlpl_b, NN, 5, 500,  false, false, false);
    gemm(stream, z2,   mlpl_w + 500 * 5,  ulog, nullptr,NN, 5, 500,  false, false, true);
    gemm(stream, z3,   mlpl_w + 1000 * 5, ulog, nullptr,NN, 5, 2000, false, false, true);
    gemm(stream, o_z4, mlpl_w + 3000 * 5, ulog, nullptr,NN, 5, 10,   false, false, true);
    gemm(stream, o_z,  mlpl_w + 3010 * 5, ulog, nullptr,NN, 5, 10,   false, false, true);
    gate_norm<<<lin_grid(NN), 256, 0, stream>>>(ulog, u, NN, 5);

    k_net_in<<<lin_grid((long)NN * 3020), 256, 0, stream>>>(u, z1, z2, z3, o_z4, o_z, netin, NN);
    gemm(stream, netin, gnnz_w, tsm,   nullptr, NN, 10, 3020, false, false, false);
    gemm(stream, adjs,  tsm,    o_net, nullptr, NN, 10, NN,   false, false, false);
    row_softmax<<<NN, 256, 0, stream>>>(o_net, o_pred, 10);

    k_x_in<<<lin_grid((long)NN * 3010), 256, 0, stream>>>(u, h1, h2, h3, o_z, xin, NN);
    gemm(stream, xin, xin, o_simx, nullptr, NN, NN, 3010, true, false, false);
    row_softmax<<<NN, 256, 0, stream>>>(o_simx, o_simx, NN);
    gemm(stream, xin, gnnh_w, tsm,   nullptr, NN, 10, 3010, false, false, false);
    gemm(stream, adj, tsm,    xoutp, nullptr, NN, 10, NN,   false, false, false);
    row_softmax<<<NN, 256, 0, stream>>>(xoutp, o_xpred, 10);

    // ---- Student-t cluster assignment (V=1) ----
    k_student_q<<<lin_grid(NN), 256, 0, stream>>>(o_zq, cluster, o_q, NN);
}
